// SambaYSelfDecoderLayer_40097814675983
// MI455X (gfx1250) — matrix operand order repaired and checked by reading, not compiled, not run on hardware
//
#include <hip/hip_runtime.h>
#include <stdint.h>

// CDNA5 / gfx1250 flash attention (packed varlen, sliding window, GQA)
// wave32, v_wmma_f32_16x16x32_bf16, bf16 K/V preconverted into d_ws,
// exp2-domain softmax (raw v_exp_f32), permlane16 max-reduce,
// softmax denominator accumulated by WMMA against a ones matrix.

typedef __attribute__((ext_vector_type(16))) __bf16 v16bf;
typedef __attribute__((ext_vector_type(8)))  float  v8f;

#define HQ   16
#define HKV  4
#define DIM  128
#define SEQ  2048
#define GQA  (HQ / HKV)
#define NEG_BIG (-1.0e30f)

static __device__ inline unsigned short f32_bf16bits(float f) {
  uint32_t u = __builtin_bit_cast(uint32_t, f);
  uint32_t r = u + 0x7fffu + ((u >> 16) & 1u);   // round-to-nearest-even
  return (unsigned short)(r >> 16);
}
static __device__ inline uint32_t pack_bf2(float a, float b) {
  return (uint32_t)f32_bf16bits(a) | ((uint32_t)f32_bf16bits(b) << 16);
}
static __device__ inline __bf16 bits_bf(unsigned short s) {
  return __builtin_bit_cast(__bf16, s);
}
static __device__ inline __bf16 f_bf(float f) { return bits_bf(f32_bf16bits(f)); }

// raw v_exp_f32 (exp2); no denormal-guard expansion — args are <= 0 here
static __device__ inline float fexp2(float x) {
#if __has_builtin(__builtin_amdgcn_exp2f)
  return __builtin_amdgcn_exp2f(x);
#else
  return exp2f(x);
#endif
}

struct B32x8 { uint4 lo, hi; };
static __device__ inline v16bf frag_from(uint4 lo, uint4 hi) {
  B32x8 t{lo, hi};
  return __builtin_bit_cast(v16bf, t);
}
static __device__ inline v16bf frag_load(const unsigned short* p) {
  const uint4* p4 = (const uint4*)p;
  return frag_from(p4[0], p4[1]);
}

// ---- 16-lane butterfly max: v_permlane16_b32 (VALU, no LDS) ----
#if __has_builtin(__builtin_amdgcn_permlane16)
static __device__ inline float permx(float v, uint32_t s0, uint32_t s1) {
  uint32_t u = __builtin_bit_cast(uint32_t, v);
  uint32_t r = __builtin_amdgcn_permlane16(u, u, s0, s1, false, false);
  return __builtin_bit_cast(float, r);
}
static __device__ inline float red_max16(float v) {
  v = fmaxf(v, permx(v, 0x67452301u, 0xEFCDAB89u));   // lane ^ 1
  v = fmaxf(v, permx(v, 0x54761032u, 0xDCFE98BAu));   // lane ^ 2
  v = fmaxf(v, permx(v, 0x32107654u, 0xBA98FEDCu));   // lane ^ 4
  v = fmaxf(v, permx(v, 0xFEDCBA98u, 0x76543210u));   // lane ^ 8
  return v;
}
#else
static __device__ inline float red_max16(float v) {
  for (int d = 1; d < 16; d <<= 1) v = fmaxf(v, __shfl_xor(v, d, 32));
  return v;
}
#endif

// ---------------- preprocessing: K -> bf16 row-major ----------------
__global__ __launch_bounds__(256) void cvt_k_kernel(const float* __restrict__ K,
                                                    uint32_t* __restrict__ Kb) {
  const int i = blockIdx.x * blockDim.x + threadIdx.x;   // pair index
  const float2 f = ((const float2*)K)[i];
  Kb[i] = pack_bf2(f.x, f.y);
}

// ---------------- preprocessing: V -> bf16 transposed [hkv][d][seq] ----------------
__global__ __launch_bounds__(256) void cvt_v_kernel(const float* __restrict__ V,
                                                    uint32_t* __restrict__ Vt) {
  const int i   = blockIdx.x * blockDim.x + threadIdx.x; // output pair index
  const int lp  = i & (SEQ / 2 - 1);                     // pair of seq positions
  const int hkd = i >> 10;                               // hk*DIM + d
  const int hk  = hkd >> 7;
  const int d   = hkd & (DIM - 1);
  const int l   = lp * 2;
  const float a = V[((size_t)l       * HKV + hk) * DIM + d];
  const float b = V[((size_t)(l + 1) * HKV + hk) * DIM + d];
  Vt[i] = pack_bf2(a, b);
}

// ---------------- main flash-attention kernel ----------------
__global__ __launch_bounds__(128, 1)
void swa_flash_kernel(const float*          __restrict__ Q,
                      const unsigned short* __restrict__ Kb,   // bf16 [seq][hkv][d]
                      const unsigned short* __restrict__ Vt,   // bf16 [hkv][d][seq]
                      const int*            __restrict__ cu,
                      const int*            __restrict__ winp,
                      float*                __restrict__ O)
{
  __shared__ unsigned short pbuf[4][16 * 32];   // per-wave P transpose buffer

  const int lane = threadIdx.x & 31;
  const int wv   = threadIdx.x >> 5;
  const int g    = lane >> 4;          // half-wave select
  const int nn   = lane & 15;          // N (and A-row) index within half
  const int h    = blockIdx.y;         // query head
  const int hk   = h / GQA;            // kv head
  const int q0   = blockIdx.x * 64 + wv * 16;

  const int   win = winp[0];
  const int   c1 = cu[1], c2 = cu[2], c3 = cu[3];
  // scale * log2(e): softmax runs in exp2 domain
  const float scale = 0.08838834764831845f * 1.4426950408889634f;

  // ---- Q A-fragments (exp2-domain scale folded in) ----
  v16bf aq[4];
  {
    const float* qrow = Q + ((size_t)(q0 + nn) * HQ + h) * DIM;
#pragma unroll
    for (int c = 0; c < 4; ++c) {
      const float4* p0 = (const float4*)(qrow + c * 32 + 16 * g);
      const float4* p1 = (const float4*)(qrow + c * 32 + 16 * g + 8);
      float4 f0 = p0[0], f1 = p0[1], f2 = p1[0], f3 = p1[1];
      float t[16] = {f0.x, f0.y, f0.z, f0.w, f1.x, f1.y, f1.z, f1.w,
                     f2.x, f2.y, f2.z, f2.w, f3.x, f3.y, f3.z, f3.w};
#pragma unroll
      for (int e = 0; e < 16; ++e) aq[c][e] = f_bf(t[e] * scale);
    }
  }

  // ---- constant ones B-fragment for denominator WMMA (P x 1) ----
  v16bf ones;
#pragma unroll
  for (int e = 0; e < 16; ++e) ones[e] = bits_bf((unsigned short)0x3F80u);

  // ---- per-row state; C/D layout: lane holds (M = r + 8g, N = nn) ----
  // same-seg & window & causal collapses to: lo[r] <= k <= qr
  int   lo[8];
  float mrun[8];
  v8f   lacc;            // softmax denominator, same layout as C/D rows
  v8f   acc[8];
#pragma unroll
  for (int r = 0; r < 8; ++r) {
    const int qr = q0 + r + 8 * g;
    const int s  = (qr >= c1) + (qr >= c2) + (qr >= c3);
    const int ss = (s == 0) ? 0 : (s == 1 ? c1 : (s == 2 ? c2 : c3));
    const int wlo = qr - win + 1;
    lo[r]   = wlo > ss ? wlo : ss;
    mrun[r] = NEG_BIG;
    lacc[r] = 0.0f;
  }
#pragma unroll
  for (int dd = 0; dd < 8; ++dd)
#pragma unroll
    for (int r = 0; r < 8; ++r) acc[dd][r] = 0.0f;

  // ---- key range for this q-tile ----
  const int sq0      = (q0 >= c1) + (q0 >= c2) + (q0 >= c3);
  const int segstart = (sq0 == 0) ? 0 : (sq0 == 1 ? c1 : (sq0 == 2 ? c2 : c3));
  int kmin = q0 - win + 1;
  if (kmin < segstart) kmin = segstart;
  if (kmin < 0) kmin = 0;
  const int kmax = q0 + 15;   // keys touched <= (kmax & ~31) + 31 <= SEQ-1

  for (int kt = (kmin & ~31); kt <= kmax; kt += 32) {
    // ---- S = Q * K^T for 16q x 32k tile (8 WMMAs) ----
    const size_t kbase0 = ((size_t)(kt + nn)      * HKV + hk) * DIM + 16 * g;
    const size_t kbase1 = ((size_t)(kt + 16 + nn) * HKV + hk) * DIM + 16 * g;

    v8f s0, s1;
#pragma unroll
    for (int r = 0; r < 8; ++r) { s0[r] = 0.0f; s1[r] = 0.0f; }

#pragma unroll
    for (int c = 0; c < 4; ++c) {
      v16bf bk0 = frag_load(Kb + kbase0 + c * 32);
      s0 = __builtin_amdgcn_wmma_f32_16x16x32_bf16(false, aq[c], false, bk0,
                                                   (short)0, s0, false, false);
      v16bf bk1 = frag_load(Kb + kbase1 + c * 32);
      s1 = __builtin_amdgcn_wmma_f32_16x16x32_bf16(false, aq[c], false, bk1,
                                                   (short)0, s1, false, false);
    }

    if (kt + 32 <= kmax) {   // gfx1250 global_prefetch_b8 on next tiles
      __builtin_prefetch(Kb + ((size_t)(kt + 32 + nn) * HKV + hk) * DIM, 0, 0);
      __builtin_prefetch(Vt + ((size_t)(hk * DIM) + nn) * SEQ + kt + 32, 0, 0);
    }

    // ---- mask + online softmax (rows live in 16-lane halves) ----
    const int k0 = kt + nn, k1 = kt + 16 + nn;
    float alpha[8];
#pragma unroll
    for (int r = 0; r < 8; ++r) {
      const int qr = q0 + r + 8 * g;
      const float v0 = (k0 >= lo[r] && k0 <= qr) ? s0[r] : NEG_BIG;
      const float v1 = (k1 >= lo[r] && k1 <= qr) ? s1[r] : NEG_BIG;

      const float mx = red_max16(fmaxf(v0, v1));
      const float mo = mrun[r];
      const float mn = fmaxf(mo, mx);
      const float al = fexp2(mo - mn);          // single v_exp_f32
      const float p0 = fexp2(v0 - mn);
      const float p1 = fexp2(v1 - mn);

      mrun[r]  = mn;
      alpha[r] = al;

      // P tile (bf16 bits) to LDS, row-major [m][k], for A-layout re-read
      pbuf[wv][(r + 8 * g) * 32 + nn]      = f32_bf16bits(p0);
      pbuf[wv][(r + 8 * g) * 32 + 16 + nn] = f32_bf16bits(p1);
    }

#if __has_builtin(__builtin_amdgcn_s_wait_dscnt)
    __builtin_amdgcn_s_wait_dscnt(0);
#endif

    // ---- re-read P as A-fragment (16x32, K-dim = 32 keys) ----
    v16bf pa;
#pragma unroll
    for (int j = 0; j < 8; ++j) {
      const int kb = (j << 1) + 16 * g;  // even -> b32 load
      const uint32_t pr = *(const uint32_t*)&pbuf[wv][nn * 32 + kb];
      pa[2 * j]     = bits_bf((unsigned short)(pr & 0xffffu));
      pa[2 * j + 1] = bits_bf((unsigned short)(pr >> 16));
    }

    // ---- denominator: l = l*alpha + P x ones (WMMA, same row layout) ----
    {
      v8f cl;
#pragma unroll
      for (int r = 0; r < 8; ++r) cl[r] = lacc[r] * alpha[r];
      lacc = __builtin_amdgcn_wmma_f32_16x16x32_bf16(false, pa, false, ones,
                                                     (short)0, cl, false, false);
    }

    // ---- O = O*alpha + P*V, one WMMA per 16-wide d-subtile (8 WMMAs) ----
#pragma unroll
    for (int dd = 0; dd < 8; ++dd) {
      v16bf bv = frag_load(Vt + ((size_t)(hk * DIM) + dd * 16 + nn) * SEQ + kt + 16 * g);
      v8f cacc;
#pragma unroll
      for (int r = 0; r < 8; ++r) cacc[r] = acc[dd][r] * alpha[r];
      acc[dd] = __builtin_amdgcn_wmma_f32_16x16x32_bf16(false, pa, false, bv,
                                                        (short)0, cacc, false, false);
    }
  }

  // ---- normalize + store ----
#pragma unroll
  for (int r = 0; r < 8; ++r) {
    const float inv = 1.0f / lacc[r];
    const int   qr  = q0 + r + 8 * g;
    float* orow = O + ((size_t)qr * HQ + h) * DIM;
#pragma unroll
    for (int dd = 0; dd < 8; ++dd) orow[dd * 16 + nn] = acc[dd][r] * inv;
  }
}

extern "C" void kernel_launch(void* const* d_in, const int* in_sizes, int n_in,
                              void* d_out, int out_size, void* d_ws, size_t ws_size,
                              hipStream_t stream) {
  (void)in_sizes; (void)n_in; (void)ws_size; (void)out_size;
  const float* q   = (const float*)d_in[0];
  const float* k   = (const float*)d_in[1];
  const float* v   = (const float*)d_in[2];
  const int*   cu  = (const int*)d_in[3];
  const int*   win = (const int*)d_in[4];
  float* out = (float*)d_out;

  // workspace: Kb (2 MB bf16) then Vt (2 MB bf16)
  unsigned short* Kb = (unsigned short*)d_ws;
  unsigned short* Vt = Kb + (size_t)SEQ * HKV * DIM;

  {
    const int pairs = SEQ * HKV * DIM / 2;    // 524288
    hipLaunchKernelGGL(cvt_k_kernel, dim3(pairs / 256), dim3(256), 0, stream,
                       k, (uint32_t*)Kb);
    hipLaunchKernelGGL(cvt_v_kernel, dim3(pairs / 256), dim3(256), 0, stream,
                       v, (uint32_t*)Vt);
  }

  dim3 grid(SEQ / 64, HQ, 1);   // 32 q-blocks (64 rows = 4 waves) x 16 heads
  dim3 block(128, 1, 1);
  hipLaunchKernelGGL(swa_flash_kernel, grid, block, 0, stream,
                     q, Kb, Vt, cu, win, out);
}
